// GraphMPNEncoder_36756330119412
// MI455X (gfx1250) — matrix operand order repaired and checked by reading, not compiled, not run on hardware
//
#include <hip/hip_runtime.h>
#include <hip/hip_bf16.h>

typedef __attribute__((ext_vector_type(16))) __bf16 v16bf;
typedef __attribute__((ext_vector_type(8)))  __bf16 v8bf;
typedef __attribute__((ext_vector_type(8)))  float  v8f;

#define N_ATOMS   300000
#define N_COLS    8
#define VOCAB     4096
#define D_COL     32
#define HID       256
#define N_BONDS   320000
#define B_GRAPHS  6000
#define SENTINEL  999999999
#define NTILES    (N_ATOMS / 16)   // 18750, exact

// ---------------- f32 -> bf16 weight conversion ----------------
__global__ void cvt_bf16_kernel(const float* __restrict__ src,
                                __bf16* __restrict__ dst, int n) {
    int i = blockIdx.x * blockDim.x + threadIdx.x;
    if (i < n) dst[i] = (__bf16)src[i];
}

// ---------------- zero a buffer (16B granules) ----------------
__global__ void zero_kernel(uint4* __restrict__ p, size_t n16) {
    size_t i = (size_t)blockIdx.x * blockDim.x + threadIdx.x;
    size_t stride = (size_t)gridDim.x * blockDim.x;
    uint4 z = {0u, 0u, 0u, 0u};
    for (; i < n16; i += stride) p[i] = z;
}

// ---------------- embedding sum + input linear (WMMA) ----------------
__global__ void embed_in_kernel(const int*    __restrict__ afeat,   // [N_ATOMS][8]
                                const float*  __restrict__ emb,     // [8][VOCAB][32]
                                const __bf16* __restrict__ Wb,      // [HID][32] bf16
                                const float*  __restrict__ bias,    // [HID]
                                __bf16*       __restrict__ h)       // [N_ATOMS][HID]
{
    __shared__ float xs[8][16][33];
    const int lane = threadIdx.x & 31;
    const int wv   = threadIdx.x >> 5;
    const int tile = blockIdx.x * 8 + wv;
    if (tile >= NTILES) return;                 // wave-uniform, no block barriers here
    const int atom0 = tile * 16;

    for (int a = 0; a < 16; ++a) {
        const int atom = atom0 + a;
        float s = 0.f;
#pragma unroll
        for (int c = 0; c < N_COLS; ++c) {
            int f  = afeat[atom * N_COLS + c];
            int id = (f >= SENTINEL) ? 0 : (f & (VOCAB - 1));
            s += emb[((size_t)c * VOCAB + id) * D_COL + lane];
        }
        xs[wv][a][lane] = s;
    }

    const int row  = lane & 15;
    const int koff = (lane >= 16) ? 8 : 0;
    v16bf A;
#pragma unroll
    for (int j = 0; j < 8; ++j) {
        A[j]     = (__bf16)xs[wv][row][koff + j];
        A[j + 8] = (__bf16)xs[wv][row][16 + koff + j];
    }

    const int nlane = lane & 15;
    const int kb    = (lane >= 16) ? 8 : 0;
    const int mo    = (lane >= 16) ? 8 : 0;
#pragma unroll 1
    for (int nt = 0; nt < 16; ++nt) {
        const int n = nt * 16 + nlane;
        v8bf b0 = *(const v8bf*)(Wb + (size_t)n * D_COL + kb);
        v8bf b1 = *(const v8bf*)(Wb + (size_t)n * D_COL + kb + 16);
        v16bf B;
#pragma unroll
        for (int j = 0; j < 8; ++j) { B[j] = b0[j]; B[j + 8] = b1[j]; }
        float bv = bias[n];
        v8f c;
#pragma unroll
        for (int r = 0; r < 8; ++r) c[r] = bv;
        c = __builtin_amdgcn_wmma_f32_16x16x32_bf16(false, A, false, B,
                                                    (short)0, c, false, false);
#pragma unroll
        for (int r = 0; r < 8; ++r)
            h[(size_t)(atom0 + r + mo) * HID + nt * 16 + nlane] = (__bf16)c[r];
    }
}

// ---------------- edge scatter: agg[dst] += h[src] (packed bf16 atomics) ----------------
__global__ void scatter_kernel(const int*    __restrict__ bfeat,  // [N_BONDS][3]
                               const __bf16* __restrict__ h,
                               __bf16*       __restrict__ agg)
{
    const size_t t  = (size_t)blockIdx.x * blockDim.x + threadIdx.x;
    const int    e  = (int)(t >> 7);        // directed edge id, 128 lanes/edge
    const int    c2 = (int)(t & 127);       // packed channel pair
    int src, dst;
    if (e < N_BONDS) { src = bfeat[e * 3 + 0]; dst = bfeat[e * 3 + 1]; }
    else { int e2 = e - N_BONDS; src = bfeat[e2 * 3 + 1]; dst = bfeat[e2 * 3 + 0]; }

    unsigned val = *(const unsigned*)((const char*)h + ((size_t)src * HID + 2 * c2) * 2);
    unsigned long long addr =
        (unsigned long long)((char*)agg + ((size_t)dst * HID + 2 * c2) * 2);
    asm volatile("global_atomic_pk_add_bf16 %0, %1, off scope:SCOPE_DEV"
                 :: "v"(addr), "v"(val) : "memory");
}

// ---------------- GEMM(s) + bias (+optional second GEMM, optional ReLU) ----------------
// hout = act( hin @ W1^T + b1 [+ agg @ W2^T + b2] ), bf16 in/out, f32 WMMA accumulate.
// Weight k-slices staged in LDS per block via async global->LDS (TDM-lite path),
// shared by all 8 waves; B operands come from ds_load_b128.
template <bool HAS_AGG, bool RELU>
__global__ void __launch_bounds__(256)
mpn_kernel(const __bf16* __restrict__ hin,
           const __bf16* __restrict__ agg,
           const __bf16* __restrict__ W1, const float* __restrict__ b1,
           const __bf16* __restrict__ W2, const float* __restrict__ b2,
           __bf16*       __restrict__ hout)
{
    constexpr int NMAT = HAS_AGG ? 2 : 1;
    // [mat][n][k-slice], rows padded to 40 halves (80B, keeps 16B alignment,
    // rotates LDS banks between consecutive n)
    __shared__ __attribute__((aligned(16))) __bf16 Wlds[NMAT][HID][40];

    const int lane = threadIdx.x & 31;
    int tile = blockIdx.x * 8 + (threadIdx.x >> 5);
    if (tile >= NTILES) tile = NTILES - 1;      // keep all waves alive for barriers
    const int atom0 = tile * 16;

    const int row   = lane & 15;
    const int koffA = (lane >= 16) ? 8 : 0;
    const int nlane = lane & 15;
    const int koffB = (lane >= 16) ? 8 : 0;

    v8f acc[16];
#pragma unroll
    for (int nt = 0; nt < 16; ++nt) {
        float bv = b1[nt * 16 + nlane];
        if (HAS_AGG) bv += b2[nt * 16 + nlane];
#pragma unroll
        for (int r = 0; r < 8; ++r) acc[nt][r] = bv;
    }

    const __bf16* hrow = hin + (size_t)(atom0 + row) * HID;
    const __bf16* arow = HAS_AGG ? (agg + (size_t)(atom0 + row) * HID) : nullptr;

#pragma unroll 1
    for (int kt = 0; kt < 8; ++kt) {
        __syncthreads();   // previous slice fully consumed before overwrite
        // stage this kt-slice of the weights: NMAT * 256 rows * 32 halfs,
        // copied as 16B chunks with async global->LDS
        for (int q = threadIdx.x; q < NMAT * HID * 4; q += 256) {
            const int mat = q >> 10;            // q / (HID*4)
            const int r   = q & (HID * 4 - 1);
            const int n   = r >> 2;
            const int kp  = (r & 3) * 8;
            const __bf16* gsrc =
                ((mat == 0) ? W1 : W2) + (size_t)n * HID + kt * 32 + kp;
            unsigned ldsoff = (unsigned)(size_t)&Wlds[mat][n][kp];
            unsigned long long ga = (unsigned long long)(size_t)gsrc;
            asm volatile("global_load_async_to_lds_b128 %0, %1, off"
                         :: "v"(ldsoff), "v"(ga) : "memory");
        }
        asm volatile("s_wait_asynccnt 0x0" ::: "memory");
        __syncthreads();   // slice visible to all waves

        const int kA = kt * 32 + koffA;
        v8bf a0 = *(const v8bf*)(hrow + kA);
        v8bf a1 = *(const v8bf*)(hrow + kA + 16);
        v16bf A1;
#pragma unroll
        for (int j = 0; j < 8; ++j) { A1[j] = a0[j]; A1[j + 8] = a1[j]; }
        v16bf A2;
        if (HAS_AGG) {
            v8bf g0 = *(const v8bf*)(arow + kA);
            v8bf g1 = *(const v8bf*)(arow + kA + 16);
#pragma unroll
            for (int j = 0; j < 8; ++j) { A2[j] = g0[j]; A2[j + 8] = g1[j]; }
        }

#pragma unroll
        for (int nt = 0; nt < 16; ++nt) {
            const int n = nt * 16 + nlane;
            v8bf w0 = *(const v8bf*)&Wlds[0][n][koffB];
            v8bf w1 = *(const v8bf*)&Wlds[0][n][koffB + 16];
            v16bf B1;
#pragma unroll
            for (int j = 0; j < 8; ++j) { B1[j] = w0[j]; B1[j + 8] = w1[j]; }
            acc[nt] = __builtin_amdgcn_wmma_f32_16x16x32_bf16(
                false, A1, false, B1, (short)0, acc[nt], false, false);
            if (HAS_AGG) {
                v8bf u0 = *(const v8bf*)&Wlds[1][n][koffB];
                v8bf u1 = *(const v8bf*)&Wlds[1][n][koffB + 16];
                v16bf B2;
#pragma unroll
                for (int j = 0; j < 8; ++j) { B2[j] = u0[j]; B2[j + 8] = u1[j]; }
                acc[nt] = __builtin_amdgcn_wmma_f32_16x16x32_bf16(
                    false, A2, false, B2, (short)0, acc[nt], false, false);
            }
        }
    }

    const int mo = (lane >= 16) ? 8 : 0;
#pragma unroll
    for (int nt = 0; nt < 16; ++nt) {
#pragma unroll
        for (int r = 0; r < 8; ++r) {
            float v = acc[nt][r];
            if (RELU) v = fmaxf(v, 0.f);
            hout[(size_t)(atom0 + r + mo) * HID + nt * 16 + nlane] = (__bf16)v;
        }
    }
}

// ---------------- segment mean readout ----------------
__global__ void readout_kernel(const __bf16* __restrict__ g,      // [N_ATOMS][HID]
                               const int*    __restrict__ scopes, // [B_GRAPHS][2]
                               float*        __restrict__ out)    // [B_GRAPHS][HID]
{
    const int gi = blockIdx.x;
    const int t  = threadIdx.x;       // 256 = HID
    const int start = scopes[gi * 2 + 0];
    const int len   = scopes[gi * 2 + 1];
    float s = 0.f;
    for (int j = 0; j < len; ++j)
        s += (float)g[(size_t)(start + j) * HID + t];
    int d = (len > 0) ? len : 1;
    out[(size_t)gi * HID + t] = s / (float)d;
}

extern "C" void kernel_launch(void* const* d_in, const int* in_sizes, int n_in,
                              void* d_out, int out_size, void* d_ws, size_t ws_size,
                              hipStream_t stream) {
    const int*   afeat   = (const int*)  d_in[0];
    const int*   bfeat   = (const int*)  d_in[1];
    const int*   scopes  = (const int*)  d_in[2];
    const float* emb     = (const float*)d_in[3];
    const float* W_in    = (const float*)d_in[4];
    const float* b_in    = (const float*)d_in[5];
    const float* W_self  = (const float*)d_in[6];
    const float* b_self  = (const float*)d_in[7];
    const float* W_neigh = (const float*)d_in[8];
    const float* b_neigh = (const float*)d_in[9];
    const float* W_out   = (const float*)d_in[10];
    const float* b_out   = (const float*)d_in[11];

    char* ws = (char*)d_ws;
    const size_t HB = (size_t)N_ATOMS * HID * 2;   // one bf16 h buffer
    __bf16* h0       = (__bf16*)(ws);
    __bf16* h1       = (__bf16*)(ws + HB);
    __bf16* agg      = (__bf16*)(ws + 2 * HB);
    __bf16* Wb_in    = (__bf16*)(ws + 3 * HB);
    __bf16* Wb_self  = Wb_in   + HID * D_COL;
    __bf16* Wb_neigh = Wb_self + HID * HID;
    __bf16* Wb_out   = Wb_neigh + HID * HID;

    cvt_bf16_kernel<<<(HID * D_COL + 255) / 256, 256, 0, stream>>>(W_in,   Wb_in,   HID * D_COL);
    cvt_bf16_kernel<<<(HID * HID + 255) / 256, 256, 0, stream>>>(W_self,  Wb_self,  HID * HID);
    cvt_bf16_kernel<<<(HID * HID + 255) / 256, 256, 0, stream>>>(W_neigh, Wb_neigh, HID * HID);
    cvt_bf16_kernel<<<(HID * HID + 255) / 256, 256, 0, stream>>>(W_out,   Wb_out,   HID * HID);

    const int mblocks = (NTILES + 7) / 8;          // 8 waves (tiles) per block
    embed_in_kernel<<<mblocks, 256, 0, stream>>>(afeat, emb, Wb_in, b_in, h0);

    __bf16* cur = h0;
    __bf16* nxt = h1;
    const size_t aggN16 = HB / 16;
    const int sblocks = (2 * N_BONDS * 128) / 256; // 320000 blocks
    for (int s = 0; s < 3; ++s) {
        zero_kernel<<<4096, 256, 0, stream>>>((uint4*)agg, aggN16);
        scatter_kernel<<<sblocks, 256, 0, stream>>>(bfeat, cur, agg);
        mpn_kernel<true, true><<<mblocks, 256, 0, stream>>>(
            cur, agg, Wb_self, b_self, Wb_neigh, b_neigh, nxt);
        __bf16* t = cur; cur = nxt; nxt = t;
    }

    // g = h @ W_out^T + b_out  (no relu), stored bf16 in the free buffer
    mpn_kernel<false, false><<<mblocks, 256, 0, stream>>>(
        cur, nullptr, Wb_out, b_out, nullptr, nullptr, nxt);

    readout_kernel<<<B_GRAPHS, 256, 0, stream>>>(nxt, scopes, (float*)d_out);
}
